// TimeSlicedRelationalKGRUL_81793357185094
// MI455X (gfx1250) — compile-verified, matrix operand order checked by reading, not run on hardware
//
#include <hip/hip_runtime.h>
#include <math.h>

#define Bc   8
#define Tc   512
#define Dc   128
#define NHc  8
#define DFFc 256
#define Nc   2048
#define Ec   32768
#define Rc   8
#define Mc   256
#define EPSc 1e-5f

typedef __attribute__((ext_vector_type(16))) _Float16 v16h;
typedef __attribute__((ext_vector_type(8)))  float    v8f;

// ---------------------------------------------------------------------------
// WMMA fragment helpers (wave32, 16x16x32 f16->f32).  Layouts per CDNA5 ISA
// 7.12.2: A operand = (M=16 x K=32) row striped per lane; B operand has the
// same per-lane layout interpreted as (N=16 x K=32); C/D: col = lane&15,
// rows (lane>=16 ? 8..15 : 0..7) in v0..v7.
// ---------------------------------------------------------------------------
__device__ __forceinline__ v16h frag_rm(const float* base, int ld, int kmax) {
  int lane = threadIdx.x & 31;
  int row  = lane & 15;
  int hi   = (lane >> 4) << 3;
  const float* r = base + (long)row * ld;
  v16h f;
#pragma unroll
  for (int i = 0; i < 8; ++i) {
    int k0 = hi + i, k1 = 16 + hi + i;
    f[i]     = (k0 < kmax) ? (_Float16)r[k0] : (_Float16)0.0f;
    f[i + 8] = (k1 < kmax) ? (_Float16)r[k1] : (_Float16)0.0f;
  }
  return f;
}

// B operand sourced from a (K x N)-stored matrix (transposed load)
__device__ __forceinline__ v16h frag_cm(const float* base, int ld, int kmax) {
  int lane = threadIdx.x & 31;
  int n    = lane & 15;
  int hi   = (lane >> 4) << 3;
  v16h f;
#pragma unroll
  for (int i = 0; i < 8; ++i) {
    int k0 = hi + i, k1 = 16 + hi + i;
    f[i]     = (k0 < kmax) ? (_Float16)base[(long)k0 * ld + n] : (_Float16)0.0f;
    f[i + 8] = (k1 < kmax) ? (_Float16)base[(long)k1 * ld + n] : (_Float16)0.0f;
  }
  return f;
}

__device__ __forceinline__ void store_c(float* base, int ld, v8f c) {
  int lane = threadIdx.x & 31;
  int col  = lane & 15;
  int mb   = (lane >> 4) << 3;
#pragma unroll
  for (int r = 0; r < 8; ++r) base[(long)(mb + r) * ld + col] = c[r];
}

__device__ __forceinline__ v8f wmma_f16(v16h a, v16h b, v8f c) {
  return __builtin_amdgcn_wmma_f32_16x16x32_f16(false, a, false, b, (short)0, c,
                                                false, false);
}

// ---------------------------------------------------------------------------
// Generic GEMM: C[M,N] = act(A[M,K] @ W[N,K]^T + bias[N]).
// One wave per 64x16 strip: the W (B-operand) fragment is loaded/converted
// once per K-step and reused across 4 stacked M tiles (4 WMMAs per step).
// global_prefetch of the next K tile overlaps L2 latency with the WMMAs.
// ---------------------------------------------------------------------------
template <int RELU>
__global__ __launch_bounds__(32) void k_gemm(const float* __restrict__ A, int lda,
                                             const float* __restrict__ W, int ldw,
                                             const float* __restrict__ bias,
                                             float* __restrict__ C, int ldc, int K) {
  int mt = blockIdx.x, nt = blockIdx.y;
  const float* Ab = A + (long)mt * 64 * lda;
  const float* Wb = W + (long)nt * 16 * ldw;
  int lane = threadIdx.x & 31;
  int col  = lane & 15;
  int prow = lane & 15;
  float bv = bias ? bias[nt * 16 + col] : 0.0f;
  v8f acc[4];
#pragma unroll
  for (int m = 0; m < 4; ++m)
#pragma unroll
    for (int r = 0; r < 8; ++r) acc[m][r] = bv;

  for (int k0 = 0; k0 < K; k0 += 32) {
    if (k0 + 32 < K) {  // prefetch next K tile (A strip rows + W rows)
      __builtin_prefetch(Wb + (long)prow * ldw + k0 + 32, 0, 3);
#pragma unroll
      for (int m = 0; m < 4; ++m)
        __builtin_prefetch(Ab + (long)(m * 16 + prow) * lda + k0 + 32, 0, 3);
    }
    v16h b = frag_rm(Wb + k0, ldw, 32);
#pragma unroll
    for (int m = 0; m < 4; ++m) {
      v16h a = frag_rm(Ab + (long)m * 16 * lda + k0, lda, 32);
      acc[m] = wmma_f16(a, b, acc[m]);
    }
  }
#pragma unroll
  for (int m = 0; m < 4; ++m) {
    if (RELU) {
#pragma unroll
      for (int r = 0; r < 8; ++r) acc[m][r] = fmaxf(acc[m][r], 0.0f);
    }
    store_c(C + ((long)mt * 64 + m * 16) * ldc + nt * 16, ldc, acc[m]);
  }
}

// ---------------------------------------------------------------------------
// Attention: one wave handles 16 query rows of one (b,h).  Scores strip kept
// in LDS, softmax in fp32, both GEMMs via WMMA.
// ---------------------------------------------------------------------------
template <int DH, int TK>
__global__ __launch_bounds__(32) void k_attn(const float* __restrict__ Qb, int qRow, long qBat,
                                             const float* __restrict__ Kb, int kRow, long kBat,
                                             const float* __restrict__ Vb, int vRow, long vBat,
                                             float* __restrict__ Ob, int oRow, long oBat,
                                             float scale) {
  __shared__ float s[16 * TK];
  __shared__ float rowM[16], rowS[16];
  int qt = blockIdx.x, h = blockIdx.y, b = blockIdx.z;
  const float* Q  = Qb + (long)b * qBat + (long)qt * 16 * qRow + h * DH;
  const float* Km = Kb + (long)b * kBat + h * DH;
  const float* Vm = Vb + (long)b * vBat + h * DH;
  float*       O  = Ob + (long)b * oBat + (long)qt * 16 * oRow + h * DH;
  int lane = threadIdx.x;
  int col = lane & 15, mb = (lane >> 4) << 3;

  v16h aq = frag_rm(Q, qRow, DH);
  for (int kt = 0; kt < TK / 16; ++kt) {
    v16h bk = frag_rm(Km + (long)kt * 16 * kRow, kRow, DH);
    v8f c;
#pragma unroll
    for (int r = 0; r < 8; ++r) c[r] = 0.0f;
    c = wmma_f16(aq, bk, c);
#pragma unroll
    for (int r = 0; r < 8; ++r) s[(mb + r) * TK + kt * 16 + col] = c[r] * scale;
  }
  __syncthreads();
  if (lane < 16) {
    float m = -1e30f;
    for (int j = 0; j < TK; ++j) m = fmaxf(m, s[lane * TK + j]);
    float sum = 0.0f;
    for (int j = 0; j < TK; ++j) sum += expf(s[lane * TK + j] - m);
    rowM[lane] = m;
    rowS[lane] = 1.0f / sum;
  }
  __syncthreads();
  for (int idx = lane; idx < 16 * TK; idx += 32) {
    int rr = idx / TK;
    s[idx] = expf(s[idx] - rowM[rr]) * rowS[rr];
  }
  __syncthreads();

  v8f oacc[DH / 16];
#pragma unroll
  for (int nt = 0; nt < DH / 16; ++nt)
#pragma unroll
    for (int r = 0; r < 8; ++r) oacc[nt][r] = 0.0f;
  for (int kt = 0; kt < TK / 32; ++kt) {
    v16h ap = frag_rm(&s[kt * 32], TK, 32);
#pragma unroll
    for (int nt = 0; nt < DH / 16; ++nt) {
      v16h bv = frag_cm(Vm + (long)kt * 32 * vRow + nt * 16, vRow, 32);
      oacc[nt] = wmma_f16(ap, bv, oacc[nt]);
    }
  }
#pragma unroll
  for (int nt = 0; nt < DH / 16; ++nt) store_c(O + nt * 16, oRow, oacc[nt]);
}

// ---------------------------------------------------------------------------
// LayerNorm kernels: MODE 0 -> LN(x + y), MODE 1 -> LN(relu(x)).  1 wave/row.
// ---------------------------------------------------------------------------
template <int MODE>
__global__ __launch_bounds__(32) void k_ln(const float* __restrict__ x,
                                           const float* __restrict__ y,
                                           const float* __restrict__ g,
                                           const float* __restrict__ bb,
                                           float* __restrict__ out) {
  long row = blockIdx.x;
  int lane = threadIdx.x;
  const float* xr = x + row * Dc;
  float v[4];
  float sum = 0.0f;
#pragma unroll
  for (int i = 0; i < 4; ++i) {
    int d = lane + 32 * i;
    float t = xr[d];
    if (MODE == 0) t += y[row * Dc + d];
    else t = fmaxf(t, 0.0f);
    v[i] = t;
    sum += t;
  }
#pragma unroll
  for (int off = 16; off; off >>= 1) sum += __shfl_xor(sum, off, 32);
  float mu = sum * (1.0f / Dc);
  float var = 0.0f;
#pragma unroll
  for (int i = 0; i < 4; ++i) { float dlt = v[i] - mu; var += dlt * dlt; }
#pragma unroll
  for (int off = 16; off; off >>= 1) var += __shfl_xor(var, off, 32);
  var *= (1.0f / Dc);
  float inv = rsqrtf(var + EPSc);
#pragma unroll
  for (int i = 0; i < 4; ++i) {
    int d = lane + 32 * i;
    out[row * Dc + d] = (v[i] - mu) * inv * g[d] + bb[d];
  }
}

// z += sinusoidal positional encoding
__global__ void k_pe(float* __restrict__ z) {
  long idx = (long)blockIdx.x * blockDim.x + threadIdx.x;
  if (idx >= (long)Bc * Tc * Dc) return;
  int d = idx & (Dc - 1);
  int t = (idx / Dc) % Tc;
  int i2 = (d >> 1) << 1;
  float freq = expf(-(float)i2 * (logf(10000.0f) / Dc));
  float ang = (float)t * freq;
  z[idx] += (d & 1) ? cosf(ang) : sinf(ang);
}

// h += type_emb[type_id]
__global__ void k_type(float* __restrict__ h, const int* __restrict__ tid,
                       const float* __restrict__ emb) {
  long idx = (long)blockIdx.x * blockDim.x + threadIdx.x;
  if (idx >= (long)Bc * Nc * Dc) return;
  long row = idx >> 7;
  int d = idx & (Dc - 1);
  h[idx] += emb[(long)tid[row] * Dc + d];
}

// one wave per edge: msg = W_rel[rid] @ h[src]; scatter-add to agg/deg
__global__ __launch_bounds__(256) void k_edges(const int* __restrict__ ei,
                                               const int* __restrict__ et,
                                               const float* __restrict__ h,
                                               const float* __restrict__ wrel,
                                               float* __restrict__ agg,
                                               float* __restrict__ deg, int rid) {
  __shared__ float hs[8][Dc];
  int wid = threadIdx.x >> 5, lane = threadIdx.x & 31;
  long eg = (long)blockIdx.x * 8 + wid;
  int b = (int)(eg / Ec), e = (int)(eg % Ec);
  if (et[(long)b * Ec + e] != rid) return;
  int s = ei[(long)b * 2 * Ec + e];
  int dn = ei[(long)b * 2 * Ec + Ec + e];
  const float* hsrc = h + ((long)b * Nc + s) * Dc;
  for (int i = lane; i < Dc; i += 32) hs[wid][i] = hsrc[i];
  float* ag = agg + ((long)b * Nc + dn) * Dc;
  for (int o = lane; o < Dc; o += 32) {
    const float* wr = wrel + (long)o * Dc;
    float acc = 0.0f;
    for (int k = 0; k < Dc; ++k) acc += wr[k] * hs[wid][k];
    atomicAdd(&ag[o], acc);
  }
  if (lane == 0) atomicAdd(&deg[(long)b * Nc + dn], 1.0f);
}

__global__ void k_combine(float* __restrict__ out, const float* __restrict__ agg,
                          const float* __restrict__ deg) {
  long idx = (long)blockIdx.x * blockDim.x + threadIdx.x;
  if (idx >= (long)Bc * Nc * Dc) return;
  out[idx] += agg[idx] / fmaxf(deg[idx >> 7], 1.0f);
}

__global__ void k_gather_mem(float* __restrict__ mem, const float* __restrict__ h,
                             const int* __restrict__ midx) {
  long idx = (long)blockIdx.x * blockDim.x + threadIdx.x;
  if (idx >= (long)Bc * Mc * Dc) return;
  long bm = idx >> 7;
  int b = (int)(bm / Mc), d = (int)(idx & (Dc - 1));
  mem[idx] = h[((long)b * Nc + midx[bm]) * Dc + d];
}

__global__ void k_gather_tgt(float* __restrict__ tgt, const float* __restrict__ h,
                             const int* __restrict__ tix) {
  long idx = (long)blockIdx.x * blockDim.x + threadIdx.x;
  if (idx >= (long)Bc * Dc) return;
  int b = (int)(idx >> 7), d = (int)(idx & (Dc - 1));
  tgt[idx] = h[((long)b * Nc + tix[b]) * Dc + d];
}

__global__ __launch_bounds__(128) void k_mean(const float* __restrict__ z,
                                              float* __restrict__ out) {
  int b = blockIdx.x, d = threadIdx.x;
  float s = 0.0f;
  for (int t = 0; t < Tc; ++t) s += z[((long)b * Tc + t) * Dc + d];
  out[b * Dc + d] = s * (1.0f / Tc);
}

// joint = [seq, tgt] (256) -> 128 relu -> OUT
template <int OUT>
__global__ __launch_bounds__(128) void k_head(const float* __restrict__ seq,
                                              const float* __restrict__ tgt,
                                              const float* __restrict__ w1,
                                              const float* __restrict__ b1,
                                              const float* __restrict__ w2,
                                              const float* __restrict__ b2,
                                              float* __restrict__ out) {
  __shared__ float hid[128];
  int b = blockIdx.x, j = threadIdx.x;
  const float* sp = seq + b * Dc;
  const float* tp = tgt + b * Dc;
  const float* w = w1 + (long)j * 256;
  float acc = b1[j];
  for (int k = 0; k < 128; ++k) acc += w[k] * sp[k];
  for (int k = 0; k < 128; ++k) acc += w[128 + k] * tp[k];
  hid[j] = fmaxf(acc, 0.0f);
  __syncthreads();
  if (j < OUT) {
    float o = b2[j];
    const float* w2r = w2 + (long)j * 128;
    for (int k = 0; k < 128; ++k) o += w2r[k] * hid[k];
    out[b * OUT + j] = o;
  }
}

// ---------------------------------------------------------------------------
extern "C" void kernel_launch(void* const* d_in, const int* in_sizes, int n_in,
                              void* d_out, int out_size, void* d_ws, size_t ws_size,
                              hipStream_t stream) {
  (void)in_sizes; (void)n_in; (void)out_size; (void)ws_size;
  const float* x_seq    = (const float*)d_in[0];
  const float* nodef    = (const float*)d_in[1];
  const int*   ntype    = (const int*)d_in[2];
  const int*   eindex   = (const int*)d_in[3];
  const int*   etype    = (const int*)d_in[4];
  const int*   midx     = (const int*)d_in[5];
  const int*   tix      = (const int*)d_in[6];
  // params in JAX pytree (sorted-key) order
  const float* e_ffn_b1 = (const float*)d_in[7];
  const float* e_ffn_b2 = (const float*)d_in[8];
  const float* e_ffn_w1 = (const float*)d_in[9];
  const float* e_ffn_w2 = (const float*)d_in[10];
  const float* e_ln1_b  = (const float*)d_in[11];
  const float* e_ln1_g  = (const float*)d_in[12];
  const float* e_ln2_b  = (const float*)d_in[13];
  const float* e_ln2_g  = (const float*)d_in[14];
  const float* e_out_b  = (const float*)d_in[15];
  const float* e_out_w  = (const float*)d_in[16];
  const float* e_qkv_b  = (const float*)d_in[17];
  const float* e_qkv_w  = (const float*)d_in[18];
  const float* feat_b   = (const float*)d_in[19];
  const float* feat_w   = (const float*)d_in[20];
  const float* f_ln_b   = (const float*)d_in[21];
  const float* f_ln_g   = (const float*)d_in[22];
  const float* f_out_b  = (const float*)d_in[23];
  const float* f_out_w  = (const float*)d_in[24];
  const float* f_qkv_b  = (const float*)d_in[25];
  const float* f_qkv_w  = (const float*)d_in[26];
  const float* g_ln_b   = (const float*)d_in[27];
  const float* g_ln_g   = (const float*)d_in[28];
  const float* g_rel_w  = (const float*)d_in[29];
  const float* g_self_b = (const float*)d_in[30];
  const float* g_self_w = (const float*)d_in[31];
  const float* in_b     = (const float*)d_in[32];
  const float* in_w     = (const float*)d_in[33];
  const float* r_b1     = (const float*)d_in[34];
  const float* r_b2     = (const float*)d_in[35];
  const float* r_w1     = (const float*)d_in[36];
  const float* r_w2     = (const float*)d_in[37];
  const float* s_b1     = (const float*)d_in[38];
  const float* s_b2     = (const float*)d_in[39];
  const float* s_w1     = (const float*)d_in[40];
  const float* s_w2     = (const float*)d_in[41];
  const float* type_emb = (const float*)d_in[42];

  float* dout = (float*)d_out;
  float* seqo = dout + 8 + 32;          // seq_summary (B x 128)
  float* tgto = dout + 8 + 32 + 1024;   // target      (B x 128)

  float* ws = (float*)d_ws;
  float* z    = ws; ws += (long)Bc * Tc * Dc;        // 4096 x 128
  float* qkv  = ws; ws += (long)Bc * Tc * 3 * Dc;    // 4096 x 384
  float* abuf = ws; ws += (long)Bc * Tc * Dc;        // 4096 x 128
  float* tbuf = ws; ws += (long)Bc * Tc * DFFc;      // 4096 x 256
  float* h    = ws; ws += (long)Bc * Nc * Dc;        // 16384 x 128
  float* hout = ws; ws += (long)Bc * Nc * Dc;
  float* agg  = ws; ws += (long)Bc * Nc * Dc;
  float* deg  = ws; ws += (long)Bc * Nc;
  float* memb = ws; ws += (long)Bc * Mc * Dc;        // 2048 x 128
  float* kvb  = ws; ws += (long)Bc * Mc * 2 * Dc;    // 2048 x 256
  float* qbuf = ws; ws += (long)Bc * Tc * Dc;

  const int MT = Bc * Tc;   // 4096 sequence rows
  const int MG = Bc * Nc;   // 16384 node rows

  // ---- input projection + positional encoding ----
  k_gemm<0><<<dim3(MT / 64, Dc / 16), 32, 0, stream>>>(x_seq, 32, in_w, 32, in_b, z, Dc, 32);
  k_pe<<<(MT * Dc) / 256, 256, 0, stream>>>(z);

  // ---- transformer encoder ----
  for (int l = 0; l < 3; ++l) {
    k_gemm<0><<<dim3(MT / 64, 384 / 16), 32, 0, stream>>>(
        z, Dc, e_qkv_w + (long)l * 384 * Dc, Dc, e_qkv_b + l * 384, qkv, 384, Dc);
    k_attn<16, Tc><<<dim3(Tc / 16, NHc, Bc), 32, 0, stream>>>(
        qkv, 384, (long)Tc * 384, qkv + 128, 384, (long)Tc * 384,
        qkv + 256, 384, (long)Tc * 384, abuf, Dc, (long)Tc * Dc, 0.25f);
    k_gemm<0><<<dim3(MT / 64, Dc / 16), 32, 0, stream>>>(
        abuf, Dc, e_out_w + (long)l * Dc * Dc, Dc, e_out_b + l * Dc, tbuf, Dc, Dc);
    k_ln<0><<<MT, 32, 0, stream>>>(z, tbuf, e_ln1_g + l * Dc, e_ln1_b + l * Dc, z);
    k_gemm<1><<<dim3(MT / 64, DFFc / 16), 32, 0, stream>>>(
        z, Dc, e_ffn_w1 + (long)l * DFFc * Dc, Dc, e_ffn_b1 + l * DFFc, tbuf, DFFc, Dc);
    k_gemm<0><<<dim3(MT / 64, Dc / 16), 32, 0, stream>>>(
        tbuf, DFFc, e_ffn_w2 + (long)l * Dc * DFFc, DFFc, e_ffn_b2 + l * Dc, abuf, Dc, DFFc);
    k_ln<0><<<MT, 32, 0, stream>>>(z, abuf, e_ln2_g + l * Dc, e_ln2_b + l * Dc, z);
  }

  // ---- graph: feature projection + type embedding ----
  k_gemm<0><<<dim3(MG / 64, Dc / 16), 32, 0, stream>>>(nodef, 32, feat_w, 32, feat_b, h, Dc, 32);
  k_type<<<(MG * Dc) / 256, 256, 0, stream>>>(h, ntype, type_emb);

  // ---- relational graph conv, 2 layers x 8 relations ----
  for (int li = 0; li < 2; ++li) {
    k_gemm<0><<<dim3(MG / 64, Dc / 16), 32, 0, stream>>>(
        h, Dc, g_self_w + (long)li * Dc * Dc, Dc, g_self_b + li * Dc, hout, Dc, Dc);
    for (int rid = 0; rid < Rc; ++rid) {
      hipMemsetAsync(agg, 0, (size_t)MG * Dc * sizeof(float), stream);
      hipMemsetAsync(deg, 0, (size_t)MG * sizeof(float), stream);
      k_edges<<<(Bc * Ec) / 8, 256, 0, stream>>>(
          eindex, etype, h, g_rel_w + ((long)li * Rc + rid) * Dc * Dc, agg, deg, rid);
      k_combine<<<(MG * Dc) / 256, 256, 0, stream>>>(hout, agg, deg);
    }
    k_ln<1><<<MG, 32, 0, stream>>>(hout, nullptr, g_ln_g + li * Dc, g_ln_b + li * Dc, h);
  }

  // ---- gathers ----
  k_gather_tgt<<<(Bc * Dc) / 256 + 1, 256, 0, stream>>>(tgto, h, tix);
  k_gather_mem<<<(Bc * Mc * Dc) / 256, 256, 0, stream>>>(memb, h, midx);

  // ---- fusion cross-attention (nh = 4, dh = 32) ----
  k_gemm<0><<<dim3(MT / 64, Dc / 16), 32, 0, stream>>>(z, Dc, f_qkv_w, Dc, f_qkv_b, qbuf, Dc, Dc);
  k_gemm<0><<<dim3((Bc * Mc) / 64, (2 * Dc) / 16), 32, 0, stream>>>(
      memb, Dc, f_qkv_w + 128 * Dc, Dc, f_qkv_b + 128, kvb, 2 * Dc, Dc);
  k_attn<32, Mc><<<dim3(Tc / 16, 4, Bc), 32, 0, stream>>>(
      qbuf, Dc, (long)Tc * Dc, kvb, 2 * Dc, (long)Mc * 2 * Dc,
      kvb + 128, 2 * Dc, (long)Mc * 2 * Dc, abuf, Dc, (long)Tc * Dc,
      0.1767766953f /* 1/sqrt(32) */);
  k_gemm<0><<<dim3(MT / 64, Dc / 16), 32, 0, stream>>>(abuf, Dc, f_out_w, Dc, f_out_b, tbuf, Dc, Dc);
  k_ln<0><<<MT, 32, 0, stream>>>(z, tbuf, f_ln_g, f_ln_b, z);

  // ---- summary + heads ----
  k_mean<<<Bc, 128, 0, stream>>>(z, seqo);
  k_head<1><<<Bc, 128, 0, stream>>>(seqo, tgto, r_w1, r_b1, r_w2, r_b2, dout);
  k_head<4><<<Bc, 128, 0, stream>>>(seqo, tgto, s_w1, s_b1, s_w2, s_b2, dout + 8);
}
